// VectorQuantizer_Supervised_70729521431111
// MI455X (gfx1250) — matrix operand order repaired and checked by reading, not compiled, not run on hardware
//
#include <hip/hip_runtime.h>
#include <cstdint>
#include <math.h>

// ---------------------------------------------------------------------------
// VQ-VAE vector quantizer for MI455X (gfx1250, wave32, WMMA).
//   X: [8192, 4096] f32, codebook: [512, 4096] f32
//   out = [ loss(1) | quantized(8192*4096) | perplexity(1) | encodings(8192*512) ]
// Scores via bf16 WMMA (f32 accumulate), argmin per row (N-split over 2 blocks
// for occupancy), exact f32 gather for quantized/loss. Workspace ~4.2 MB.
// ---------------------------------------------------------------------------

typedef __attribute__((ext_vector_type(16))) __bf16 v16bf;
typedef __attribute__((ext_vector_type(8)))  float  v8f;

union FragBF { uint4 u4[2]; v16bf bf; };

#define D_DIM   4096
#define E_DIM   512
#define B_DIM   8192
#define M_TILE  64
#define KC      64          // K elements per LDS chunk (two K=32 WMMA steps)
#define A_STRIDE 36         // LDS row stride in dwords (padded, 16B aligned)

__device__ __forceinline__ uint32_t pk_bf16(float a, float b) {
  uint32_t ua = __float_as_uint(a);
  uint32_t ub = __float_as_uint(b);
  ua = (ua + 0x7FFFu + ((ua >> 16) & 1u)) >> 16;   // round-to-nearest-even
  ub = (ub + 0x7FFFu + ((ub >> 16) & 1u)) >> 16;
  return (ub << 16) | (ua & 0xFFFFu);
}

__device__ __forceinline__ void cvt16_store(const float4* __restrict__ src,
                                            uint32_t* __restrict__ dst_lds) {
  float4 f0 = src[0], f1 = src[1], f2 = src[2], f3 = src[3];
  uint4 lo, hi;
  lo.x = pk_bf16(f0.x, f0.y); lo.y = pk_bf16(f0.z, f0.w);
  lo.z = pk_bf16(f1.x, f1.y); lo.w = pk_bf16(f1.z, f1.w);
  hi.x = pk_bf16(f2.x, f2.y); hi.y = pk_bf16(f2.z, f2.w);
  hi.z = pk_bf16(f3.x, f3.y); hi.w = pk_bf16(f3.z, f3.w);
  *reinterpret_cast<uint4*>(dst_lds)     = lo;
  *reinterpret_cast<uint4*>(dst_lds + 4) = hi;
}

// --- Pack codebook into WMMA B-fragment layout (bf16), per (kstep, ntile) ---
// B 32x16 bf16, this code's contiguous k order: lanes 0-15 -> K=0..15 (VGPR v holds K=2v,2v+1),
// lanes 16-31 -> K=16..31. N = lane%16. Fragment = 32 lanes x 8 dwords = 1KB.
__global__ __launch_bounds__(256) void pack_codebook(
    const float* __restrict__ cb, uint32_t* __restrict__ Bpk) {
  const int f     = blockIdx.x * 8 + (threadIdx.x >> 5);  // 0..4095 fragments
  const int lane  = threadIdx.x & 31;
  const int kstep = f >> 5;                               // 0..127 (K/32)
  const int ntile = f & 31;                               // 0..31  (E/16)
  const int n     = ntile * 16 + (lane & 15);
  const int kofs  = kstep * 32 + ((lane >> 4) & 1) * 16;
  const float4* src = reinterpret_cast<const float4*>(cb + (size_t)n * D_DIM + kofs);
  float4 f0 = src[0], f1 = src[1], f2 = src[2], f3 = src[3];
  uint4 lo, hi;
  lo.x = pk_bf16(f0.x, f0.y); lo.y = pk_bf16(f0.z, f0.w);
  lo.z = pk_bf16(f1.x, f1.y); lo.w = pk_bf16(f1.z, f1.w);
  hi.x = pk_bf16(f2.x, f2.y); hi.y = pk_bf16(f2.z, f2.w);
  hi.z = pk_bf16(f3.x, f3.y); hi.w = pk_bf16(f3.z, f3.w);
  uint32_t* dst = Bpk + ((size_t)f * 32 + lane) * 8;
  *reinterpret_cast<uint4*>(dst)     = lo;
  *reinterpret_cast<uint4*>(dst + 4) = hi;
}

// --- Codebook row squared norms ---
__global__ __launch_bounds__(256) void codebook_norms(
    const float* __restrict__ cb, float* __restrict__ nrm) {
  const int e = blockIdx.x, tid = threadIdx.x;
  float s = 0.f;
  for (int j = tid; j < D_DIM; j += 256) {
    float v = cb[(size_t)e * D_DIM + j];
    s += v * v;
  }
  __shared__ float red[256];
  red[tid] = s; __syncthreads();
  for (int o = 128; o > 0; o >>= 1) {
    if (tid < o) red[tid] += red[tid + o];
    __syncthreads();
  }
  if (tid == 0) nrm[e] = red[0];
}

// --- Main: scores = ||e||^2 - 2 x.e via bf16 WMMA, partial argmin ---
// Grid 256: block b -> rows [(b>>1)*64, +64), codes [(b&1)*256, +256).
// Adjacent blocks share the same X tile (L2/WGP$ reuse).
__global__ __launch_bounds__(256) void vq_gemm_argmin(
    const float* __restrict__ X, const uint32_t* __restrict__ Bpk,
    const float* __restrict__ cbnorm, float* __restrict__ pval,
    int* __restrict__ pidx) {
  __shared__ uint32_t Atile[2][M_TILE * A_STRIDE];  // bf16 pairs as dwords
  __shared__ float red_val[M_TILE][8];
  __shared__ int   red_idx[M_TILE][8];

  const int tid   = threadIdx.x;
  const int lane  = tid & 31;
  const int wv    = tid >> 5;          // wave 0..7 owns 32 cols (2 n-tiles)
  const int half  = (lane >> 4) & 1;
  const int l16   = lane & 15;
  const int bid   = blockIdx.x;
  const int m0    = (bid >> 1) * M_TILE;
  const int nhalf = bid & 1;           // which 256-code half
  const int ntile0 = nhalf * 16 + wv * 2;       // first of this wave's 2 n-tiles
  const int ncol0  = ntile0 * 16 + l16;         // lane's column in n-tile 0

  // A staging: 64 rows x 64 k of f32 -> bf16 LDS, 16 floats per thread
  const int s_row = tid >> 2;
  const int s_cg  = tid & 3;
  const float* srcA = X + (size_t)(m0 + s_row) * D_DIM + s_cg * 16;
  uint32_t* dstA0 = &Atile[0][s_row * A_STRIDE + s_cg * 8];
  uint32_t* dstA1 = &Atile[1][s_row * A_STRIDE + s_cg * 8];

  const v8f vzero = {0.f, 0.f, 0.f, 0.f, 0.f, 0.f, 0.f, 0.f};
  v8f acc[4][2];
  #pragma unroll
  for (int i = 0; i < 4; ++i)
    #pragma unroll
    for (int j = 0; j < 2; ++j) acc[i][j] = vzero;

  cvt16_store(reinterpret_cast<const float4*>(srcA), dstA0);

  for (int ch = 0; ch < D_DIM / KC; ++ch) {
    __syncthreads();
    const int buf = ch & 1;
    if (ch + 1 < D_DIM / KC)
      cvt16_store(reinterpret_cast<const float4*>(srcA + (ch + 1) * KC),
                  buf ? dstA0 : dstA1);

    #pragma unroll
    for (int ks = 0; ks < 2; ++ks) {
      const int kstep = ch * 2 + ks;   // global K/32 step, 0..127
      FragBF bfr[2];
      #pragma unroll
      for (int nt = 0; nt < 2; ++nt) {
        const uint32_t* bs =
            Bpk + ((size_t)(kstep * 32 + ntile0 + nt) * 32 + lane) * 8;
        bfr[nt].u4[0] = *reinterpret_cast<const uint4*>(bs);
        bfr[nt].u4[1] = *reinterpret_cast<const uint4*>(bs + 4);
      }
      // Load ALL four A fragments into distinct registers first so the
      // ds_loads can be clause-grouped and drained progressively while
      // earlier WMMAs execute (avoids one-load-in-flight serialization).
      FragBF af[4];
      #pragma unroll
      for (int ms = 0; ms < 4; ++ms) {
        // A 16x32 frag, contiguous k order as packed B: lanes 0-15 M=lane, v0..7 K=0..15;
        // lanes 16-31 M=lane-16, v0..7 K=16..31.
        const uint32_t* as =
            &Atile[buf][(ms * 16 + l16) * A_STRIDE + ks * 16 + half * 8];
        af[ms].u4[0] = *reinterpret_cast<const uint4*>(as);
        af[ms].u4[1] = *reinterpret_cast<const uint4*>(as + 4);
      }
      #pragma unroll
      for (int ms = 0; ms < 4; ++ms)
        #pragma unroll
        for (int nt = 0; nt < 2; ++nt)
          acc[ms][nt] = __builtin_amdgcn_wmma_f32_16x16x32_bf16(
              false, af[ms].bf, false, bfr[nt].bf, (short)0, acc[ms][nt],
              false, false);
    }
  }

  // Epilogue: dist proxy = ||e||^2 - 2*dot; argmin per row over this half.
  float nrm[2];
  #pragma unroll
  for (int nt = 0; nt < 2; ++nt) nrm[nt] = cbnorm[ncol0 + nt * 16];

  #pragma unroll
  for (int ms = 0; ms < 4; ++ms) {
    #pragma unroll
    for (int r = 0; r < 8; ++r) {
      float bv = 3.4e38f; int bi = 0x7fffffff;
      #pragma unroll
      for (int nt = 0; nt < 2; ++nt) {
        float v = nrm[nt] - 2.0f * acc[ms][nt][r];
        int n = ncol0 + nt * 16;
        if (v < bv || (v == bv && n < bi)) { bv = v; bi = n; }
      }
      // reduce across the 16 lanes of this half-group (cols)
      #pragma unroll
      for (int off = 8; off >= 1; off >>= 1) {
        float ov = __shfl_xor(bv, off, 32);
        int   oi = __shfl_xor(bi, off, 32);
        if (ov < bv || (ov == bv && oi < bi)) { bv = ov; bi = oi; }
      }
      if (l16 == 0) {
        const int ml = ms * 16 + half * 8 + r;   // C tile: lanes>=16 -> M+8
        red_val[ml][wv] = bv;
        red_idx[ml][wv] = bi;
      }
    }
  }
  __syncthreads();
  if (tid < M_TILE) {
    float bv = red_val[tid][0]; int bi = red_idx[tid][0];
    #pragma unroll
    for (int w = 1; w < 8; ++w) {
      float ov = red_val[tid][w]; int oi = red_idx[tid][w];
      if (ov < bv || (ov == bv && oi < bi)) { bv = ov; bi = oi; }
    }
    pval[(size_t)nhalf * B_DIM + m0 + tid] = bv;
    pidx[(size_t)nhalf * B_DIM + m0 + tid] = bi;
  }
}

// --- Merge the two N-half partial argmins (first-min tie semantics) ---
__global__ __launch_bounds__(256) void vq_argmin_merge(
    const float* __restrict__ pval, const int* __restrict__ pidx,
    int* __restrict__ indices) {
  const int m = blockIdx.x * 256 + threadIdx.x;
  float v0 = pval[m];            int i0 = pidx[m];
  float v1 = pval[B_DIM + m];    int i1 = pidx[B_DIM + m];
  // half-0 indices are always smaller: strict < keeps first-min on ties
  indices[m] = (v1 < v0) ? i1 : i0;
}

// --- Gather exact f32 codebook rows, per-row sq-error partials, one-hot ---
__global__ __launch_bounds__(256) void vq_gather(
    const float* __restrict__ X, const float* __restrict__ cb,
    const int* __restrict__ idxs, float* __restrict__ outQ,
    float* __restrict__ outEnc, float* __restrict__ blocksum) {
  const int b = blockIdx.x, tid = threadIdx.x;
  const int e = idxs[b];
  const float* q = cb + (size_t)e * D_DIM;
  const float* x = X + (size_t)b * D_DIM;
  float* oq = outQ + (size_t)b * D_DIM;   // offset +1 float => scalar access
  float s = 0.f;
  for (int j = tid; j < D_DIM; j += 256) {
    float qv = q[j];
    float d = qv - x[j];
    oq[j] = qv;
    s += d * d;
  }
  float* enc = outEnc + (size_t)b * E_DIM;
  for (int j = tid; j < E_DIM; j += 256) enc[j] = (j == e) ? 1.0f : 0.0f;

  __shared__ float red[256];
  red[tid] = s; __syncthreads();
  for (int o = 128; o > 0; o >>= 1) {
    if (tid < o) red[tid] += red[tid + o];
    __syncthreads();
  }
  if (tid == 0) blocksum[b] = red[0];
}

// --- Per-code usage counts (atomic-free, deterministic) ---
__global__ __launch_bounds__(256) void vq_counts(
    const int* __restrict__ idxs, uint32_t* __restrict__ counts) {
  const int e = blockIdx.x, tid = threadIdx.x;
  int c = 0;
  for (int i = tid; i < B_DIM; i += 256) c += (idxs[i] == e) ? 1 : 0;
  __shared__ int red[256];
  red[tid] = c; __syncthreads();
  for (int o = 128; o > 0; o >>= 1) {
    if (tid < o) red[tid] += red[tid + o];
    __syncthreads();
  }
  if (tid == 0) counts[e] = (uint32_t)red[0];
}

// --- Finalize: loss scalar + perplexity ---
__global__ __launch_bounds__(512) void vq_finalize(
    const float* __restrict__ blocksum, const uint32_t* __restrict__ counts,
    float* __restrict__ out_loss, float* __restrict__ out_perp) {
  __shared__ float red[512];
  __shared__ float lsum;
  const int tid = threadIdx.x;
  float s = 0.f;
  for (int i = tid; i < B_DIM; i += 512) s += blocksum[i];
  red[tid] = s; __syncthreads();
  for (int o = 256; o > 0; o >>= 1) {
    if (tid < o) red[tid] += red[tid + o];
    __syncthreads();
  }
  if (tid == 0) lsum = red[0];
  __syncthreads();
  float p = (float)counts[tid] * (1.0f / (float)B_DIM);
  red[tid] = p * logf(p + 1e-10f);
  __syncthreads();
  for (int o = 256; o > 0; o >>= 1) {
    if (tid < o) red[tid] += red[tid + o];
    __syncthreads();
  }
  if (tid == 0) {
    // q_latent_loss == e_latent_loss numerically -> loss = 1.25 * MSE
    *out_loss = 1.25f * lsum / (float)((size_t)B_DIM * D_DIM);
    *out_perp = expf(-red[0]);
  }
}

extern "C" void kernel_launch(void* const* d_in, const int* in_sizes, int n_in,
                              void* d_out, int out_size, void* d_ws, size_t ws_size,
                              hipStream_t stream) {
  (void)in_sizes; (void)n_in; (void)out_size; (void)ws_size;
  const float* X  = (const float*)d_in[0];   // [8192, 8,8,8,8] -> [8192,4096]
  const float* CB = (const float*)d_in[1];   // [512, 4096]
  float* out = (float*)d_out;

  const size_t QOFF = 1;
  const size_t PERP = 1 + (size_t)B_DIM * D_DIM;     // 33554433
  const size_t ENC  = PERP + 1;                      // 33554434

  // Workspace layout (dword offsets):
  uint32_t* ws32     = (uint32_t*)d_ws;
  uint32_t* Bpk      = ws32;                          // 1048576 dw (4 MB)
  float*    cbnorm   = (float*)(ws32 + 1048576);      // 512
  int*      indices  = (int*)(ws32 + 1049088);        // 8192
  uint32_t* counts   = ws32 + 1057280;                // 512
  float*    blocksum = (float*)(ws32 + 1057792);      // 8192
  float*    pval     = (float*)(ws32 + 1065984);      // 2*8192
  int*      pidx     = (int*)(ws32 + 1082368);        // 2*8192

  pack_codebook <<<512,  256, 0, stream>>>(CB, Bpk);
  codebook_norms<<<512,  256, 0, stream>>>(CB, cbnorm);
  vq_gemm_argmin<<<(B_DIM / M_TILE) * 2, 256, 0, stream>>>(X, Bpk, cbnorm,
                                                           pval, pidx);
  vq_argmin_merge<<<B_DIM / 256, 256, 0, stream>>>(pval, pidx, indices);
  vq_gather     <<<B_DIM, 256, 0, stream>>>(X, CB, indices, out + QOFF,
                                            out + ENC, blocksum);
  vq_counts     <<<512,  256, 0, stream>>>(indices, counts);
  vq_finalize   <<<1,    512, 0, stream>>>(blocksum, counts, out, out + PERP);
}
